// MambaMerge_42314017800459
// MI455X (gfx1250) — compile-verified, matrix operand order checked by reading, not run on hardware
//
#include <hip/hip_runtime.h>

typedef __attribute__((ext_vector_type(16))) _Float16 v16h;
typedef __attribute__((ext_vector_type(8)))  _Float16 v8h;
typedef __attribute__((ext_vector_type(8)))  float    v8f;

#define BATCH   4
#define CIN     128
#define LSEQ    1024
#define MROWS   (BATCH*LSEQ)      // 4096
#define DMODEL  256
#define DINNER  512
#define DSTATE  64
#define DTRANK  16
#define XDBLN   144               // DTRANK + 2*DSTATE

#define WMMA_F16(a, b, c) \
    __builtin_amdgcn_wmma_f32_16x16x32_f16(false, (a), false, (b), (short)0, (c), false, false)

// ---------------------------------------------------------------------------
// f32 -> f16 weight conversion
// ---------------------------------------------------------------------------
__global__ void k_cvt_f16(const float* __restrict__ src, _Float16* __restrict__ dst, int n) {
    int i = blockIdx.x * blockDim.x + threadIdx.x;
    if (i < n) dst[i] = (_Float16)src[i];
}

// Gather x (B,Cin,H,W) -> row-major f16 (MROWS, CIN), m = b*L + (h*W+w)
__global__ void k_pack_x(const float* __restrict__ x, _Float16* __restrict__ dst) {
    int i = blockIdx.x * blockDim.x + threadIdx.x;   // m*CIN + c
    int c = i & (CIN - 1);
    int m = i >> 7;
    int b = m >> 10;
    int l = m & (LSEQ - 1);
    dst[i] = (_Float16)x[((size_t)(b * CIN + c)) * LSEQ + l];
}

// ---------------------------------------------------------------------------
// WMMA fragment loader: 16 rows x 32 K of f16, row-major with leading dim ld.
// Matches CDNA5 16-bit A layout (B stored as (N,K) row-major is symmetric).
// lane l: row = l&15; halves [0..7] at K = k0+8*(l>>4), halves [8..15] at +16.
// ---------------------------------------------------------------------------
__device__ __forceinline__ v16h load_tile(const _Float16* __restrict__ base,
                                          int row0, int k0, int ld) {
    const int lane = threadIdx.x & 31;
    const int r    = lane & 15;
    const int hi   = lane >> 4;
    const _Float16* p = base + (size_t)(row0 + r) * ld + k0 + 8 * hi;
    v8h lo = *(const v8h*)(p);
    v8h hh = *(const v8h*)(p + 16);
    v16h f;
#pragma unroll
    for (int i = 0; i < 8; ++i) { f[i] = lo[i]; f[8 + i] = hh[i]; }
    return f;
}

// Store one 16x16 f32 C/D tile (8 VGPRs/lane layout).
__device__ __forceinline__ void store_tile(float* __restrict__ C, int m0, int n0,
                                           int ldc, const v8f& acc) {
    const int lane = threadIdx.x & 31;
    const int n = lane & 15, hi = lane >> 4;
#pragma unroll
    for (int v = 0; v < 8; ++v)
        C[(size_t)(m0 + v + 8 * hi) * ldc + n0 + n] = acc[v];
}

// ---------------------------------------------------------------------------
// Register-blocked GEMM: each wave computes a 32x32 output block (2x2 WMMA
// tiles) -> 4 WMMAs per k-step from 4 fragment loads (2x better operand reuse
// than 1 tile/wave). Requires M%32==0, N%32==0, K%32==0.
// ---------------------------------------------------------------------------
__global__ void k_wmma_gemm32(const _Float16* __restrict__ A, const _Float16* __restrict__ B,
                              float* __restrict__ C, int M, int N, int K,
                              int lda, int ldb, int ldc) {
    const int wave = (blockIdx.x * blockDim.x + threadIdx.x) >> 5;
    const int tn   = N >> 5;
    const int tm   = wave / tn;
    const int tc   = wave % tn;
    if (tm >= (M >> 5)) return;           // wave-uniform guard (grids are exact)
    const int m0 = tm << 5, n0 = tc << 5;

    v8f acc00 = {}, acc01 = {}, acc10 = {}, acc11 = {};
    for (int k0 = 0; k0 < K; k0 += 32) {
        v16h a0 = load_tile(A, m0,      k0, lda);
        v16h a1 = load_tile(A, m0 + 16, k0, lda);
        v16h b0 = load_tile(B, n0,      k0, ldb);
        v16h b1 = load_tile(B, n0 + 16, k0, ldb);
        acc00 = WMMA_F16(a0, b0, acc00);
        acc01 = WMMA_F16(a0, b1, acc01);
        acc10 = WMMA_F16(a1, b0, acc10);
        acc11 = WMMA_F16(a1, b1, acc11);
    }
    store_tile(C, m0,      n0,      ldc, acc00);
    store_tile(C, m0,      n0 + 16, ldc, acc01);
    store_tile(C, m0 + 16, n0,      ldc, acc10);
    store_tile(C, m0 + 16, n0 + 16, ldc, acc11);
}

// ---------------------------------------------------------------------------
// Generic 16x16-tile GEMM (used for N=144, which has an odd tile count).
// ---------------------------------------------------------------------------
__global__ void k_wmma_gemm16(const _Float16* __restrict__ A, const _Float16* __restrict__ B,
                              float* __restrict__ C, int M, int N, int K,
                              int lda, int ldb, int ldc) {
    const int wave = (blockIdx.x * blockDim.x + threadIdx.x) >> 5;
    const int tn   = N >> 4;
    const int tm   = wave / tn;
    const int tc   = wave % tn;
    if (tm >= (M >> 4)) return;
    const int m0 = tm << 4, n0 = tc << 4;

    v8f acc = {};
    for (int k0 = 0; k0 < K; k0 += 32) {
        v16h a = load_tile(A, m0, k0, lda);
        v16h b = load_tile(B, n0, k0, ldb);
        acc = WMMA_F16(a, b, acc);
    }
    store_tile(C, m0, n0, ldc, acc);
}

// ---------------------------------------------------------------------------
// Final GEMM (32x32 blocked): out = y_h @ out_proj^T + x_seq, scattered to
// (B, DMODEL, L) NCHW layout.
// ---------------------------------------------------------------------------
__device__ __forceinline__ void store_out_tile(const v8f& acc, int m0, int n0,
                                               const float* __restrict__ xseq,
                                               float* __restrict__ out) {
    const int lane = threadIdx.x & 31;
    const int n  = n0 + (lane & 15);
    const int hi = lane >> 4;
#pragma unroll
    for (int v = 0; v < 8; ++v) {
        int m = m0 + v + 8 * hi;
        float val = acc[v] + xseq[(size_t)m * DMODEL + n];
        int b = m >> 10, l = m & (LSEQ - 1);
        out[((size_t)(b * DMODEL + n)) * LSEQ + l] = val;
    }
}

__global__ void k_wmma_gemm_out(const _Float16* __restrict__ Y, const _Float16* __restrict__ Wo,
                                const float* __restrict__ xseq, float* __restrict__ out) {
    const int wave = (blockIdx.x * blockDim.x + threadIdx.x) >> 5;
    const int tn = DMODEL >> 5;           // 8 col-blocks of 32
    const int tm = wave / tn, tc = wave % tn;
    if (tm >= (MROWS >> 5)) return;
    const int m0 = tm << 5, n0 = tc << 5;

    v8f acc00 = {}, acc01 = {}, acc10 = {}, acc11 = {};
    for (int k0 = 0; k0 < DINNER; k0 += 32) {
        v16h a0 = load_tile(Y,  m0,      k0, DINNER);
        v16h a1 = load_tile(Y,  m0 + 16, k0, DINNER);
        v16h b0 = load_tile(Wo, n0,      k0, DINNER);
        v16h b1 = load_tile(Wo, n0 + 16, k0, DINNER);
        acc00 = WMMA_F16(a0, b0, acc00);
        acc01 = WMMA_F16(a0, b1, acc01);
        acc10 = WMMA_F16(a1, b0, acc10);
        acc11 = WMMA_F16(a1, b1, acc11);
    }
    store_out_tile(acc00, m0,      n0,      xseq, out);
    store_out_tile(acc01, m0,      n0 + 16, xseq, out);
    store_out_tile(acc10, m0 + 16, n0,      xseq, out);
    store_out_tile(acc11, m0 + 16, n0 + 16, xseq, out);
}

// ---------------------------------------------------------------------------
// LayerNorm over d_model=256: one block (256 thr) per row; output f16.
// ---------------------------------------------------------------------------
__global__ void k_layernorm(const float* __restrict__ xseq, const float* __restrict__ g,
                            const float* __restrict__ bb, _Float16* __restrict__ xn) {
    __shared__ float red[DMODEL];
    const int m = blockIdx.x, t = threadIdx.x;
    float v = xseq[(size_t)m * DMODEL + t];
    red[t] = v; __syncthreads();
    for (int s = DMODEL / 2; s > 0; s >>= 1) { if (t < s) red[t] += red[t + s]; __syncthreads(); }
    float mu = red[0] * (1.f / DMODEL);
    __syncthreads();
    float d = v - mu;
    red[t] = d * d; __syncthreads();
    for (int s = DMODEL / 2; s > 0; s >>= 1) { if (t < s) red[t] += red[t + s]; __syncthreads(); }
    float r = rsqrtf(red[0] * (1.f / DMODEL) + 1e-5f);
    xn[(size_t)m * DMODEL + t] = (_Float16)(d * r * g[t] + bb[t]);
}

// ---------------------------------------------------------------------------
// Depthwise causal conv (k=4) along L + SiLU. u lives in xz[:, 0:DINNER].
// One thread per (m, d); consecutive threads = consecutive d (coalesced).
// ---------------------------------------------------------------------------
__global__ void k_conv_silu(const float* __restrict__ xz, const float* __restrict__ cw,
                            const float* __restrict__ cb, float* __restrict__ uf,
                            _Float16* __restrict__ uh) {
    int i = blockIdx.x * blockDim.x + threadIdx.x;   // m*DINNER + d
    int d = i & (DINNER - 1);
    int m = i >> 9;
    int l = m & (LSEQ - 1);
    const float* col = xz + (size_t)m * (2 * DINNER) + d;
    float acc = cb[d];
#pragma unroll
    for (int t = 0; t < 4; ++t) {
        int lp = l - 3 + t;
        if (lp >= 0) acc += cw[d * 4 + t] * col[(long)(lp - l) * (2 * DINNER)];
    }
    float s = acc / (1.f + __expf(-acc));
    uf[i] = s;
    uh[i] = (_Float16)s;
}

// delta = softplus(dt @ dt_proj^T + b): tiny K=16 dot per output, scalar is fine.
__global__ void k_delta(const float* __restrict__ xdbl, const float* __restrict__ w,
                        const float* __restrict__ bias, float* __restrict__ delta) {
    int i = blockIdx.x * blockDim.x + threadIdx.x;   // m*DINNER + d
    int d = i & (DINNER - 1);
    int m = i >> 9;
    const float* row = xdbl + (size_t)m * XDBLN;
    const float* wr  = w + d * DTRANK;
    float acc = bias[d];
#pragma unroll
    for (int r = 0; r < DTRANK; ++r) acc += row[r] * wr[r];
    delta[i] = (acc > 20.f) ? acc : __logf(1.f + __expf(acc));
}

// ---------------------------------------------------------------------------
// Selective scan: one wave32 per (b, d); lane holds states s and s+32.
// Fused epilogue: y = (scan + u*D) * silu(z), stored f16 for the final GEMM.
// ---------------------------------------------------------------------------
__global__ void k_scan(const float* __restrict__ xz, const float* __restrict__ uf,
                       const float* __restrict__ xdbl, const float* __restrict__ delta,
                       const float* __restrict__ A_log, const float* __restrict__ Dp,
                       _Float16* __restrict__ yh) {
    const int wave = (blockIdx.x * blockDim.x + threadIdx.x) >> 5;
    const int lane = threadIdx.x & 31;
    const int b = wave >> 9;                  // 512 channels per batch
    const int d = wave & (DINNER - 1);
    const int s0 = lane, s1 = lane + 32;

    const float A0 = -__expf(A_log[d * DSTATE + s0]);
    const float A1 = -__expf(A_log[d * DSTATE + s1]);
    const float Dd = Dp[d];
    float h0 = 0.f, h1 = 0.f;

    for (int l = 0; l < LSEQ; ++l) {
        const int m = (b << 10) + l;
        const float dv = delta[(size_t)m * DINNER + d];
        const float uv = uf[(size_t)m * DINNER + d];
        const float zv = xz[(size_t)m * (2 * DINNER) + DINNER + d];
        const float* xr = xdbl + (size_t)m * XDBLN;
        __builtin_prefetch(xr + XDBLN, 0, 3);          // next step's B/C row
        const float Bv0 = xr[DTRANK + s0],          Bv1 = xr[DTRANK + s1];
        const float Cv0 = xr[DTRANK + DSTATE + s0], Cv1 = xr[DTRANK + DSTATE + s1];

        const float du = dv * uv;
        h0 = __expf(dv * A0) * h0 + du * Bv0;
        h1 = __expf(dv * A1) * h1 + du * Bv1;

        float y = h0 * Cv0 + h1 * Cv1;
#pragma unroll
        for (int off = 16; off > 0; off >>= 1) y += __shfl_xor(y, off, 32);

        if (lane == 0) {
            float o = (y + uv * Dd) * (zv / (1.f + __expf(-zv)));
            yh[(size_t)m * DINNER + d] = (_Float16)o;
        }
    }
}

// ---------------------------------------------------------------------------
extern "C" void kernel_launch(void* const* d_in, const int* in_sizes, int n_in,
                              void* d_out, int out_size, void* d_ws, size_t ws_size,
                              hipStream_t stream) {
    const float* x         = (const float*)d_in[0];
    const float* w_proj    = (const float*)d_in[1];
    const float* ln_g      = (const float*)d_in[2];
    const float* ln_b      = (const float*)d_in[3];
    const float* in_proj_w = (const float*)d_in[4];
    const float* conv_w    = (const float*)d_in[5];
    const float* conv_b    = (const float*)d_in[6];
    const float* x_proj_w  = (const float*)d_in[7];
    const float* dt_proj_w = (const float*)d_in[8];
    const float* dt_proj_b = (const float*)d_in[9];
    const float* A_log     = (const float*)d_in[10];
    const float* Dvec      = (const float*)d_in[11];
    const float* out_proj  = (const float*)d_in[12];
    float* out = (float*)d_out;

    char* ws = (char*)d_ws;
    size_t off = 0;
    auto alloc = [&](size_t bytes) -> char* {
        char* p = ws + off;
        off += (bytes + 255) & ~(size_t)255;
        return p;
    };
    _Float16* wproj_h   = (_Float16*)alloc((size_t)DMODEL * CIN * 2);
    _Float16* inproj_h  = (_Float16*)alloc((size_t)2 * DINNER * DMODEL * 2);
    _Float16* xproj_h   = (_Float16*)alloc((size_t)XDBLN * DINNER * 2);
    _Float16* outproj_h = (_Float16*)alloc((size_t)DMODEL * DINNER * 2);
    _Float16* xg_h      = (_Float16*)alloc((size_t)MROWS * CIN * 2);
    float*    xseq_f    = (float*)   alloc((size_t)MROWS * DMODEL * 4);
    _Float16* xn_h      = (_Float16*)alloc((size_t)MROWS * DMODEL * 2);
    float*    xz_f      = (float*)   alloc((size_t)MROWS * 2 * DINNER * 4);
    float*    u_f       = (float*)   alloc((size_t)MROWS * DINNER * 4);
    _Float16* u_h       = (_Float16*)alloc((size_t)MROWS * DINNER * 2);
    float*    xdbl_f    = (float*)   alloc((size_t)MROWS * XDBLN * 4);
    float*    delta_f   = (float*)   alloc((size_t)MROWS * DINNER * 4);
    _Float16* y_h       = (_Float16*)alloc((size_t)MROWS * DINNER * 2);

    auto cvt = [&](const float* s, _Float16* d, int n) {
        k_cvt_f16<<<(n + 255) / 256, 256, 0, stream>>>(s, d, n);
    };
    cvt(w_proj,    wproj_h,   DMODEL * CIN);
    cvt(in_proj_w, inproj_h,  2 * DINNER * DMODEL);
    cvt(x_proj_w,  xproj_h,   XDBLN * DINNER);
    cvt(out_proj,  outproj_h, DMODEL * DINNER);

    k_pack_x<<<(MROWS * CIN) / 256, 256, 0, stream>>>(x, xg_h);

    // GEMM1: x_seq = Xg @ w_proj^T (4096x256, K=128). 128*8=1024 waves -> 256 blocks
    k_wmma_gemm32<<<256, 128, 0, stream>>>(xg_h, wproj_h, xseq_f,
                                           MROWS, DMODEL, CIN, CIN, CIN, DMODEL);
    k_layernorm<<<MROWS, DMODEL, 0, stream>>>(xseq_f, ln_g, ln_b, xn_h);

    // GEMM2: xz = xn @ in_proj^T (4096x1024, K=256). 128*32=4096 waves -> 1024 blocks
    k_wmma_gemm32<<<1024, 128, 0, stream>>>(xn_h, inproj_h, xz_f,
                                            MROWS, 2 * DINNER, DMODEL,
                                            DMODEL, DMODEL, 2 * DINNER);

    k_conv_silu<<<(MROWS * DINNER) / 256, 256, 0, stream>>>(xz_f, conv_w, conv_b, u_f, u_h);

    // GEMM3: x_dbl = u @ x_proj^T (4096x144, K=512). 256*9=2304 waves -> 576 blocks
    k_wmma_gemm16<<<576, 128, 0, stream>>>(u_h, xproj_h, xdbl_f,
                                           MROWS, XDBLN, DINNER, DINNER, DINNER, XDBLN);

    k_delta<<<(MROWS * DINNER) / 256, 256, 0, stream>>>(xdbl_f, dt_proj_w, dt_proj_b, delta_f);

    // Scan: 2048 waves (one per (b,d)), 8 waves per block -> 256 blocks
    k_scan<<<256, 256, 0, stream>>>(xz_f, u_f, xdbl_f, delta_f, A_log, Dvec, y_h);

    // GEMM5 + residual + NCHW scatter. 128*8=1024 waves -> 256 blocks
    k_wmma_gemm_out<<<256, 128, 0, stream>>>(y_h, outproj_h, xseq_f, out);

    (void)in_sizes; (void)n_in; (void)out_size; (void)ws_size;
}
